// AttentionDecoderClassLogits_47253230190930
// MI455X (gfx1250) — matrix operand order repaired and checked by reading, not compiled, not run on hardware
//
#include <hip/hip_runtime.h>
#include <hip/hip_bf16.h>
#include <stdint.h>

#define BB    128
#define TT    20
#define DD    512
#define HH    512
#define VV    32000
#define NCLS  1000
#define TOPK  5
#define BN_EPS 1e-5f

typedef __attribute__((ext_vector_type(16))) __bf16       v16bf;
typedef __attribute__((ext_vector_type(8)))  float        v8f;
typedef __attribute__((ext_vector_type(4)))  unsigned int u32x4;
typedef __attribute__((ext_vector_type(4)))  int          i32x4;
typedef __attribute__((ext_vector_type(8)))  int          i32x8;

// ---------------------------------------------------------------------------
// TDM: one-shot DMA of a 16x32 f32 tile (row stride = lda elements) into LDS.
// D# built per CDNA5 ISA §8.3/§8.4: count=1, type=2 ("image"), data_size=4B,
// tile_dim0=32, tile_dim1=16, tensor_dim0_stride=lda. Issued wave-uniformly.
// ---------------------------------------------------------------------------
__device__ __forceinline__ void tdm_load_tile16x32(unsigned lds_byte_addr,
                                                   const float* gsrc,
                                                   int lda_elems) {
  unsigned long long ga = (unsigned long long)(uintptr_t)gsrc;
  u32x4 g0;
  g0[0] = 1u;                                   // count=1 (valid), user mode
  g0[1] = lds_byte_addr;                        // lds_addr
  g0[2] = (unsigned)ga;                         // global_addr[31:0]
  g0[3] = (unsigned)((ga >> 32) & 0x01ffffffu)  // global_addr[56:32]
          | (2u << 30);                         // type=2
  i32x8 g1;
  g1[0] = (int)(2u << 16);                      // data_size=2 (4 bytes)
  g1[1] = (int)(32u << 16);                     // tensor_dim0[15:0]=32
  g1[2] = (int)(16u << 16);                     // tensor_dim0 hi=0; tensor_dim1 lo=16
  g1[3] = (int)(32u << 16);                     // tensor_dim1 hi=0; tile_dim0=32
  g1[4] = 16;                                   // tile_dim1=16; tile_dim2=0
  g1[5] = lda_elems;                            // tensor_dim0_stride lo32
  g1[6] = 0;                                    // stride hi16; dim1_stride lo16
  g1[7] = 0;
  i32x4 zg = {0, 0, 0, 0};
#if __clang_major__ >= 23
  i32x8 zg2 = {0, 0, 0, 0, 0, 0, 0, 0};
  __builtin_amdgcn_tensor_load_to_lds(g0, g1, zg, zg, zg2, 0);
#else
  __builtin_amdgcn_tensor_load_to_lds(g0, g1, zg, zg, 0);
#endif
}

// ---------------------------------------------------------------------------
// Staging + wait, specialized at compile time (FLAGS&1: TDM, else async-LDS)
// ---------------------------------------------------------------------------
template <int FLAGS>
__device__ __forceinline__ void gemm_stage(const float* A0, const float* A1,
                                           int lda, int ksplit, int m0,
                                           int wave, int tid, int arow, int aseg,
                                           int kbase, unsigned lds_off,
                                           unsigned sbase) {
  if (FLAGS & 1) {
    if (wave == 0) {
      const float* src = (kbase < ksplit)
          ? (A0 + (long)m0 * lda + kbase)
          : (A1 + (long)m0 * lda + (kbase - ksplit));
      tdm_load_tile16x32(sbase, src, lda);
    }
  } else {
    if (tid < 128) {
      int kseg = kbase + aseg * 4;
      const float* src = (kseg < ksplit)
          ? (A0 + (long)(m0 + arow) * lda + kseg)
          : (A1 + (long)(m0 + arow) * lda + (kseg - ksplit));
      asm volatile("global_load_async_to_lds_b128 %0, %1, off"
                   :: "v"(lds_off), "v"(src) : "memory");
    }
  }
}

template <int FLAGS>
__device__ __forceinline__ void gemm_wait() {
  if (FLAGS & 1) __builtin_amdgcn_s_wait_tensorcnt(0);
  else asm volatile("s_wait_asynccnt 0" ::: "memory");
}

// One 16x32 K-slab: A fragment from LDS (bf16-converted), two B fragments
// from global (4x b128 each), two WMMAs accumulating two 16x16 N-tiles.
__device__ __forceinline__ void gemm_compute_slab(const float* __restrict__ sa,
                                                  int rowL, int kbaseA, int kbaseB,
                                                  const float* __restrict__ Bw,
                                                  long col0, long col1,
                                                  int K, int kb,
                                                  v8f& acc0, v8f& acc1) {
  v16bf a;
#pragma unroll
  for (int v = 0; v < 8; ++v) {
    int ka = kbaseA + 2 * v;
    a[2 * v]     = (__bf16)sa[rowL * 32 + ka];
    a[2 * v + 1] = (__bf16)sa[rowL * 32 + ka + 1];
  }

  const float* wrow0 = Bw + col0 * K + kb + kbaseB;
  const float* wrow1 = Bw + col1 * K + kb + kbaseB;
  __builtin_prefetch(wrow0 + 32, 0, 1);   // next K-slab (speculative)
  float4 p0 = *(const float4*)(wrow0 + 0);
  float4 p1 = *(const float4*)(wrow0 + 4);
  float4 p2 = *(const float4*)(wrow0 + 8);
  float4 p3 = *(const float4*)(wrow0 + 12);
  float4 q0 = *(const float4*)(wrow1 + 0);
  float4 q1 = *(const float4*)(wrow1 + 4);
  float4 q2 = *(const float4*)(wrow1 + 8);
  float4 q3 = *(const float4*)(wrow1 + 12);
  v16bf w0, w1;
  w0[0]=(__bf16)p0.x; w0[1]=(__bf16)p0.y; w0[2]=(__bf16)p0.z; w0[3]=(__bf16)p0.w;
  w0[4]=(__bf16)p1.x; w0[5]=(__bf16)p1.y; w0[6]=(__bf16)p1.z; w0[7]=(__bf16)p1.w;
  w0[8]=(__bf16)p2.x; w0[9]=(__bf16)p2.y; w0[10]=(__bf16)p2.z; w0[11]=(__bf16)p2.w;
  w0[12]=(__bf16)p3.x; w0[13]=(__bf16)p3.y; w0[14]=(__bf16)p3.z; w0[15]=(__bf16)p3.w;
  w1[0]=(__bf16)q0.x; w1[1]=(__bf16)q0.y; w1[2]=(__bf16)q0.z; w1[3]=(__bf16)q0.w;
  w1[4]=(__bf16)q1.x; w1[5]=(__bf16)q1.y; w1[6]=(__bf16)q1.z; w1[7]=(__bf16)q1.w;
  w1[8]=(__bf16)q2.x; w1[9]=(__bf16)q2.y; w1[10]=(__bf16)q2.z; w1[11]=(__bf16)q2.w;
  w1[12]=(__bf16)q3.x; w1[13]=(__bf16)q3.y; w1[14]=(__bf16)q3.z; w1[15]=(__bf16)q3.w;

  acc0 = __builtin_amdgcn_wmma_f32_16x16x32_bf16(
      false, a, false, w0, (short)0, acc0, false, false);
  acc1 = __builtin_amdgcn_wmma_f32_16x16x32_bf16(
      false, a, false, w1, (short)0, acc1, false, false);
}

// ---------------------------------------------------------------------------
// Generic NT GEMM via bf16 WMMA (16x16x32), f32 accumulate:
//   C[m,n] = sum_k Afeat(m,k) * Bw[n*K + k] + bias[n]
// K-split A: k < ksplit reads A0[m*lda+k], else A1[m*lda+(k-ksplit)].
// Block = 256 threads = 8 waves covering 16(M) x 256(N); each wave owns two
// 16x16 tiles sharing one A fragment. The 16x32 f32 A-slab is double-buffered
// in LDS, staged by TDM (FLAGS&1) or async global->LDS, with the next slab's
// DMA issued right after the top barrier so it overlaps WMMA work. The loop
// is unrolled 2x so buffer addresses are compile-time (requires K%64==0).
// FLAGS&2 streams C with non-temporal stores. Grid: (N/256, M/16).
// ---------------------------------------------------------------------------
template <int FLAGS>
__global__ void gemm_nt_wmma(const float* __restrict__ A0,
                             const float* __restrict__ A1,
                             int lda, int ksplit, int K,
                             const float* __restrict__ Bw,
                             const float* __restrict__ bias,
                             float* __restrict__ C, long ldc) {
  __shared__ __align__(16) float sA[2][16 * 32];

  const int tid  = threadIdx.x;
  const int lane = tid & 31;
  const int wave = tid >> 5;
  const int m0 = blockIdx.y * 16;
  const int n0 = (blockIdx.x * 8 + wave) * 32;

  const int rowL   = lane & 15;
  const long col0  = n0 + (lane & 15);
  const long col1  = col0 + 16;
  const int kbaseA = (lane < 16) ? 0 : 16;   // A: 16x32 bf16 lane layout, contiguous k = 16h + s
  const int kbaseB = (lane < 16) ? 0 : 16;   // B: 32x16 bf16 lane layout

  // per-thread async-copy segment (threads 0..127 active)
  const int at   = tid & 127;
  const int arow = at >> 3;                  // 0..15
  const int aseg = at & 7;                   // 0..7 (4 floats each)
  const unsigned lds_off0 = (unsigned)(uintptr_t)&sA[0][arow * 32 + aseg * 4];
  const unsigned lds_off1 = (unsigned)(uintptr_t)&sA[1][arow * 32 + aseg * 4];
  const unsigned sbase0   = (unsigned)(uintptr_t)&sA[0][0];
  const unsigned sbase1   = (unsigned)(uintptr_t)&sA[1][0];

  // prologue: stage slab 0 into buffer 0
  gemm_stage<FLAGS>(A0, A1, lda, ksplit, m0, wave, tid, arow, aseg,
                    0, lds_off0, sbase0);

  v8f acc0 = {}, acc1 = {};
  for (int kb = 0; kb < K; kb += 64) {
    // ---- half 1: compute buf0 (slab kb), stage buf1 (slab kb+32) ----
    gemm_wait<FLAGS>();
    __syncthreads();
    gemm_stage<FLAGS>(A0, A1, lda, ksplit, m0, wave, tid, arow, aseg,
                      kb + 32, lds_off1, sbase1);          // kb+32 < K (K%64==0)
    gemm_compute_slab(&sA[0][0], rowL, kbaseA, kbaseB, Bw,
                      col0, col1, K, kb, acc0, acc1);

    // ---- half 2: compute buf1 (slab kb+32), stage buf0 (slab kb+64) ----
    gemm_wait<FLAGS>();
    __syncthreads();
    if (kb + 64 < K) {
      gemm_stage<FLAGS>(A0, A1, lda, ksplit, m0, wave, tid, arow, aseg,
                        kb + 64, lds_off0, sbase0);
    }
    gemm_compute_slab(&sA[1][0], rowL, kbaseA, kbaseB, Bw,
                      col0, col1, K, kb + 32, acc0, acc1);
  }

  float bv0 = bias ? bias[col0] : 0.f;
  float bv1 = bias ? bias[col1] : 0.f;
#pragma unroll
  for (int r = 0; r < 8; ++r) {
    int row = m0 + ((lane < 16) ? r : 8 + r);
    float v0 = acc0[r] + bv0;
    float v1 = acc1[r] + bv1;
    float* p0 = &C[(long)row * ldc + col0];
    float* p1 = &C[(long)row * ldc + col1];
    if (FLAGS & 2) {
      __builtin_nontemporal_store(v0, p0);
      __builtin_nontemporal_store(v1, p1);
    } else {
      *p0 = v0;
      *p1 = v1;
    }
  }
}

// ---------------------------------------------------------------------------
// K1: per-row top-5 of img_rep, then mean of the 5 class embeddings -> cls(B,D)
// ---------------------------------------------------------------------------
__global__ void topk_mean_kernel(const float* __restrict__ img,
                                 const float* __restrict__ class_emb,
                                 float* __restrict__ cls) {
  const int b   = blockIdx.x;
  const int tid = threadIdx.x;
  __shared__ float s_val[256];
  __shared__ int   s_idx[256];
  __shared__ int   chosen[TOPK];

  for (int k = 0; k < TOPK; ++k) {
    float best = -3.4e38f;
    int   bidx = NCLS;
    for (int i = tid; i < NCLS; i += 256) {
      bool skip = false;
      for (int j = 0; j < k; ++j) if (chosen[j] == i) skip = true;
      if (skip) continue;
      float v = img[(long)b * NCLS + i];
      if (v > best || (v == best && i < bidx)) { best = v; bidx = i; }
    }
    s_val[tid] = best; s_idx[tid] = bidx;
    __syncthreads();
    for (int off = 128; off > 0; off >>= 1) {
      if (tid < off) {
        float v2 = s_val[tid + off]; int i2 = s_idx[tid + off];
        if (v2 > s_val[tid] || (v2 == s_val[tid] && i2 < s_idx[tid])) {
          s_val[tid] = v2; s_idx[tid] = i2;
        }
      }
      __syncthreads();
    }
    if (tid == 0) chosen[k] = s_idx[0];
    __syncthreads();
  }

  for (int d = tid; d < DD; d += 256) {
    float acc = 0.f;
    for (int j = 0; j < TOPK; ++j) acc += class_emb[(long)chosen[j] * DD + d];
    cls[(long)b * DD + d] = acc * (1.0f / TOPK);
  }
}

// ---------------------------------------------------------------------------
// K2: batchnorm stats over batch (biased variance, torch BN train mode)
// ---------------------------------------------------------------------------
__global__ void bn_stats_kernel(const float* __restrict__ proj,
                                float* __restrict__ mu,
                                float* __restrict__ var) {
  int d = blockIdx.x, tid = threadIdx.x;       // 128 threads == BB
  float v = proj[(long)tid * DD + d];
  __shared__ float s1[128], s2[128];
  s1[tid] = v; s2[tid] = v * v;
  __syncthreads();
  for (int off = 64; off > 0; off >>= 1) {
    if (tid < off) { s1[tid] += s1[tid + off]; s2[tid] += s2[tid + off]; }
    __syncthreads();
  }
  if (tid == 0) {
    float m = s1[0] * (1.0f / BB);
    mu[d]  = m;
    var[d] = s2[0] * (1.0f / BB) - m * m;
  }
}

// ---------------------------------------------------------------------------
// K3: build teacher-forced inputs x(B,T,D): t==0 -> BN(proj), else emb gather
// ---------------------------------------------------------------------------
__global__ void build_x_kernel(const float* __restrict__ proj,
                               const float* __restrict__ mu,
                               const float* __restrict__ var,
                               const float* __restrict__ gamma,
                               const float* __restrict__ beta,
                               const float* __restrict__ emb,
                               const int*   __restrict__ targets,
                               float* __restrict__ x) {
  int bt = blockIdx.x;         // b*TT + t
  int b = bt / TT, t = bt % TT;
  for (int d = threadIdx.x; d < DD; d += blockDim.x) {
    float v;
    if (t == 0) {
      v = (proj[(long)b * DD + d] - mu[d]) * rsqrtf(var[d] + BN_EPS) * gamma[d] + beta[d];
    } else {
      int tok = targets[b * TT + (t - 1)];
      v = emb[(long)tok * DD + d];
    }
    x[(long)bt * DD + d] = v;
  }
}

__global__ void zero_kernel(float* __restrict__ p, int n) {
  int i = blockIdx.x * blockDim.x + threadIdx.x;
  if (i < n) p[i] = 0.f;
}

__global__ void transpose_kernel(const float* __restrict__ W,
                                 float* __restrict__ WT) {
  int idx = blockIdx.x * 256 + threadIdx.x;    // d*HH + e
  int d = idx / HH, e = idx % HH;
  WT[(long)e * HH + d] = W[idx];
}

// ---------------------------------------------------------------------------
// GRU gate fusion (torch order r,z,n). gi precomputed for all steps; gh for
// this step. h evolves unmasked; out[:,t,:] zeroed past real_len.
// ---------------------------------------------------------------------------
__global__ void gru_gate_kernel(const float* __restrict__ gi,   // (B*T, 3H)
                                const float* __restrict__ gh,   // (B, 3H)
                                float* __restrict__ h,          // (B, H)
                                float* __restrict__ out,        // (B, T, H)
                                const int* __restrict__ real_lens,
                                int t) {
  int idx = blockIdx.x * 256 + threadIdx.x;  // b*HH + j
  int b = idx / HH, j = idx % HH;
  const float* gib = gi + ((long)b * TT + t) * (3 * HH);
  const float* ghb = gh + (long)b * (3 * HH);
  float ir = gib[j], iz = gib[HH + j], in_ = gib[2 * HH + j];
  float hr = ghb[j], hz = ghb[HH + j], hn  = ghb[2 * HH + j];
  float hp = h[idx];
  float r = 1.f / (1.f + expf(-(ir + hr)));
  float z = 1.f / (1.f + expf(-(iz + hz)));
  float n = tanhf(in_ + r * hn);
  float hnew = (1.f - z) * n + z * hp;
  h[idx] = hnew;
  out[((long)b * TT + t) * HH + j] = (t < real_lens[b]) ? hnew : 0.f;
}

// ---------------------------------------------------------------------------
// Causal bilinear attention per sample: scores -> softmax(strict past) -> ct
// ---------------------------------------------------------------------------
__global__ void attn_ct_kernel(const float* __restrict__ out,
                               const float* __restrict__ tmp,
                               float* __restrict__ ct) {
  int b = blockIdx.x;
  __shared__ float so[TT * HH];   // 40 KB
  __shared__ float sal[TT * TT];
  for (int k = threadIdx.x; k < TT * HH; k += blockDim.x)
    so[k] = out[(long)b * TT * HH + k];
  __syncthreads();

  for (int p = threadIdx.x; p < TT * TT; p += blockDim.x) {
    int t = p / TT, s = p % TT;
    float acc;
    if (s < t) {
      const float* tr = tmp + ((long)b * TT + t) * HH;
      acc = 0.f;
      for (int d = 0; d < HH; ++d) acc += tr[d] * so[s * HH + d];
    } else {
      acc = -1e9f;
    }
    sal[p] = acc;
  }
  __syncthreads();

  if (threadIdx.x < TT) {
    int t = threadIdx.x;
    float mx = -3.4e38f;
    for (int s = 0; s < TT; ++s) mx = fmaxf(mx, sal[t * TT + s]);
    float sum = 0.f;
    for (int s = 0; s < TT; ++s) { float e = expf(sal[t * TT + s] - mx); sal[t * TT + s] = e; sum += e; }
    float inv = 1.f / sum;
    for (int s = 0; s < TT; ++s) sal[t * TT + s] = (s < t) ? sal[t * TT + s] * inv : 0.f;
  }
  __syncthreads();

  for (int p = threadIdx.x; p < TT * HH; p += blockDim.x) {
    int t = p / HH, hh = p % HH;
    float acc = 0.f;
    for (int s = 0; s < t; ++s) acc += sal[t * TT + s] * so[s * HH + hh];
    ct[(long)b * TT * HH + p] = acc;
  }
}

// ---------------------------------------------------------------------------
// Host orchestration (graph-capture safe: all work on `stream`, no allocs).
// ---------------------------------------------------------------------------
extern "C" void kernel_launch(void* const* d_in, const int* in_sizes, int n_in,
                              void* d_out, int out_size, void* d_ws, size_t ws_size,
                              hipStream_t stream) {
  (void)in_sizes; (void)n_in; (void)out_size; (void)ws_size;

  const float* img       = (const float*)d_in[0];
  const int*   targets   = (const int*)  d_in[1];
  const int*   real_lens = (const int*)  d_in[2];
  const float* class_emb = (const float*)d_in[3];
  const float* proj_W    = (const float*)d_in[4];
  const float* proj_b    = (const float*)d_in[5];
  const float* bn_gamma  = (const float*)d_in[6];
  const float* bn_beta   = (const float*)d_in[7];
  const float* emb       = (const float*)d_in[8];
  const float* Wih       = (const float*)d_in[9];
  const float* Whh       = (const float*)d_in[10];
  const float* bih       = (const float*)d_in[11];
  const float* bhh       = (const float*)d_in[12];
  const float* attnW     = (const float*)d_in[13];
  const float* vocabW    = (const float*)d_in[14];
  const float* vocabB    = (const float*)d_in[15];
  float* logits = (float*)d_out;

  float* ws      = (float*)d_ws;
  float* ws_cls  = ws;                              // B*D
  float* ws_proj = ws_cls  + BB * DD;               // B*D
  float* ws_mu   = ws_proj + BB * DD;               // D
  float* ws_var  = ws_mu   + DD;                    // D
  float* ws_x    = ws_var  + DD;                    // B*T*D
  float* ws_h    = ws_x    + (long)BB * TT * DD;    // B*H
  float* ws_out  = ws_h    + BB * HH;               // B*T*H
  float* ws_tmp  = ws_out  + (long)BB * TT * HH;    // B*T*H
  float* ws_ct   = ws_tmp  + (long)BB * TT * HH;    // B*T*H
  float* ws_gi   = ws_ct   + (long)BB * TT * HH;    // B*T*3H
  float* ws_gh   = ws_gi   + (long)BB * TT * 3 * HH;// B*3H
  float* ws_wt   = ws_gh   + (long)BB * 3 * HH;     // H*H (attn_W^T)

  // Stage 1: image -> class embedding mean
  topk_mean_kernel<<<BB, 256, 0, stream>>>(img, class_emb, ws_cls);

  // Stage 2: proj = cls @ proj_W^T + proj_b  (M=128 N=512 K=512, async staging)
  gemm_nt_wmma<0><<<dim3(DD / 256, BB / 16), 256, 0, stream>>>(
      ws_cls, ws_cls, DD, DD, DD, proj_W, proj_b, ws_proj, DD);

  // Stage 3: BN (train mode) + build GRU inputs
  bn_stats_kernel<<<DD, 128, 0, stream>>>(ws_proj, ws_mu, ws_var);
  build_x_kernel<<<BB * TT, 256, 0, stream>>>(ws_proj, ws_mu, ws_var, bn_gamma,
                                              bn_beta, emb, targets, ws_x);

  // Stage 4: GRU. Input-side gates for ALL steps hoisted out of the
  // recurrence: gi = x @ Wih^T + bih  (M=2560 N=1536 K=512, TDM staging)
  gemm_nt_wmma<1><<<dim3((3 * HH) / 256, (BB * TT) / 16), 256, 0, stream>>>(
      ws_x, ws_x, DD, DD, DD, Wih, bih, ws_gi, 3 * HH);

  zero_kernel<<<(BB * HH + 255) / 256, 256, 0, stream>>>(ws_h, BB * HH);
  for (int t = 0; t < TT; ++t) {
    // gh = h @ Whh^T + bhh  (M=128 N=1536 K=512, async staging)
    gemm_nt_wmma<0><<<dim3((3 * HH) / 256, BB / 16), 256, 0, stream>>>(
        ws_h, ws_h, HH, HH, HH, Whh, bhh, ws_gh, 3 * HH);
    gru_gate_kernel<<<(BB * HH) / 256, 256, 0, stream>>>(
        ws_gi, ws_gh, ws_h, ws_out, real_lens, t);
  }

  // Stage 5: attention. tmp = out @ attn_W via NT GEMM on attn_W^T.
  transpose_kernel<<<(HH * HH) / 256, 256, 0, stream>>>(attnW, ws_wt);
  gemm_nt_wmma<0><<<dim3(HH / 256, (BB * TT) / 16), 256, 0, stream>>>(
      ws_out, ws_out, HH, HH, HH, ws_wt, nullptr, ws_tmp, HH);
  attn_ct_kernel<<<BB, 256, 0, stream>>>(ws_out, ws_tmp, ws_ct);

  // Stage 6: logits = [out|ct] @ vocab_W^T + vocab_b
  // (M=2560 N=32000 K=1024, TDM staging + non-temporal C stream so vocab_W
  //  stays L2-resident under the 328MB logits write)
  gemm_nt_wmma<3><<<dim3(VV / 256, (BB * TT) / 16), 256, 0, stream>>>(
      ws_out, ws_ct, HH, HH, 2 * HH, vocabW, vocabB, logits, VV);
}
